// QTCN_2104533975412
// MI455X (gfx1250) — compile-verified, matrix operand order checked by reading, not run on hardware
//
#include <hip/hip_runtime.h>
#include <hip/hip_bf16.h>

// ---------------------------------------------------------------------------
// QTCN on MI455X (gfx1250, wave32)
//   Kernel 1: windowed-FC projection via v_wmma_f32_16x16x32_f16
//   Kernel 2: 8-qubit statevector sim, one wave32 per state, state in VGPRs,
//             lane-bit qubits exchanged via __shfl_xor; ZZ*YY*XX fused into a
//             single pair-exchange+phase pass (1 fetch instead of 2, 1 state
//             pass instead of 3)
//   Kernel 3: per-batch mean reduction
// ---------------------------------------------------------------------------

typedef __attribute__((ext_vector_type(16))) _Float16 v16h;
typedef __attribute__((ext_vector_type(8)))  float    v8f;

#define NB     16            // batch
#define NC     8             // channels
#define NT     384           // time steps
#define KW     16            // kernel window
#define NWIN   369           // NT - (KW-1)
#define MROWS  (NB * NWIN)   // 5904 rows (369 exact 16-row WMMA tiles)
#define KDIM   (NC * KW)     // 128
#define NQ     8             // qubits -> 256 amplitudes

// ======================= Kernel 1: WMMA projection =========================
// reduced[m, n] = sum_k feat[m, k] * fc_w[n, k] + fc_b[n]
// feat[m, c*16 + k] = x[b, c, w + k] with m = b*369 + w   (DIL = 1)
__global__ void qtcn_gemm_wmma(const float* __restrict__ x,
                               const float* __restrict__ fc_w,
                               const float* __restrict__ fc_b,
                               float* __restrict__ reduced) {
  const int lane = threadIdx.x & 31;
  const int wave = blockIdx.x * (blockDim.x >> 5) + (threadIdx.x >> 5);
  if (wave >= MROWS / 16) return;                 // uniform per wave

  const int mrow = wave * 16 + (lane & 15);       // A-operand row for this lane
  const int bidx = mrow / NWIN;
  const int widx = mrow % NWIN;
  const float* __restrict__ xrow = x + bidx * NC * NT + widx; // [c*NT + k]
  __builtin_prefetch(xrow, 0, 1);                 // global_prefetch_b8

  const int khalf = (lane >> 4) & 1;              // K-half held by this lane
  const int ncol  = lane & 15;                    // B-operand column

  // Padded-B handling without divergence: always load a valid fc_w row,
  // zero the 8 padding columns with a mask multiply after the load.
  const float colmask = (ncol < NQ) ? 1.f : 0.f;
  const float* __restrict__ brow = fc_w + (ncol & (NQ - 1)) * KDIM;

  v8f acc = {};
  #pragma unroll
  for (int kc = 0; kc < 4; ++kc) {                // K = 128 in 4 chunks of 32
    const int kbase = kc * 32;
    v16h av, bv;
    // Per the 16-bit A 16x32 layout, this lane's 16 K positions are two runs
    // of 8 consecutive K values:  kbase + grp*16 + khalf*8 + t,  t = 0..7.
    // Each run stays inside one channel (no 16-boundary crossing).
    #pragma unroll
    for (int grp = 0; grp < 2; ++grp) {
      const int k0 = kbase + grp * 16 + khalf * 8;  // first K of the run
      const int c  = k0 >> 4;                       // channel
      const int kk = k0 & 15;                       // tap offset in window
      const float* __restrict__ arun = xrow + c * NT + kk;
      #pragma unroll
      for (int t = 0; t < 8; ++t)
        av[grp * 8 + t] = (_Float16)arun[t];

      // B: same K run, 16B-aligned (row base 512B, k0 multiple of 8)
      const float4 b0 = *(const float4*)(brow + k0);
      const float4 b1 = *(const float4*)(brow + k0 + 4);
      bv[grp * 8 + 0] = (_Float16)(colmask * b0.x);
      bv[grp * 8 + 1] = (_Float16)(colmask * b0.y);
      bv[grp * 8 + 2] = (_Float16)(colmask * b0.z);
      bv[grp * 8 + 3] = (_Float16)(colmask * b0.w);
      bv[grp * 8 + 4] = (_Float16)(colmask * b1.x);
      bv[grp * 8 + 5] = (_Float16)(colmask * b1.y);
      bv[grp * 8 + 6] = (_Float16)(colmask * b1.z);
      bv[grp * 8 + 7] = (_Float16)(colmask * b1.w);
    }
    acc = __builtin_amdgcn_wmma_f32_16x16x32_f16(false, av, false, bv,
                                                 (short)0, acc, false, false);
  }

  // D layout: lanes 0-15: (M=v, N=lane); lanes 16-31: (M=v+8, N=lane-16)
  if (ncol < NQ) {
    const float bias  = fc_b[ncol];
    const int   mbase = wave * 16 + ((lane < 16) ? 0 : 8);
    #pragma unroll
    for (int v = 0; v < 8; ++v)
      reduced[(mbase + v) * NQ + ncol] = acc[v] + bias;
  }
}

// ======================= Kernel 2: statevector sim =========================
struct cpx { float x, y; };
__device__ __forceinline__ cpx cmul(cpx a, cpx b) {
  return {a.x * b.x - a.y * b.y, a.x * b.y + a.y * b.x};
}
__device__ __forceinline__ cpx cadd(cpx a, cpx b) { return {a.x + b.x, a.y + b.y}; }

// partner amplitude at global index  i ^ mask ; mask bits [2:0]=reg, [7:3]=lane
__device__ __forceinline__ void fetch_mask(const cpx (&st)[8], cpx (&oth)[8], int mask) {
  const int rm = mask & 7;
  const int lm = mask >> 3;
  #pragma unroll
  for (int r = 0; r < 8; ++r) {
    cpx v = st[r ^ rm];
    if (lm) { v.x = __shfl_xor(v.x, lm, 32); v.y = __shfl_xor(v.y, lm, 32); }
    oth[r] = v;
  }
}

__device__ __forceinline__ void apply_u3(cpx (&st)[8], int lane, int p,
                                         float th, float ph, float lm) {
  const float c  = __cosf(0.5f * th), sn = __sinf(0.5f * th);
  const cpx a00 = {c, 0.f};
  const cpx a01 = {-__cosf(lm) * sn, -__sinf(lm) * sn};
  const cpx a10 = { __cosf(ph) * sn,  __sinf(ph) * sn};
  const cpx a11 = { __cosf(ph + lm) * c, __sinf(ph + lm) * c};
  cpx oth[8];
  fetch_mask(st, oth, 1 << p);
  #pragma unroll
  for (int r = 0; r < 8; ++r) {
    const int bit = (((lane << 3) | r) >> p) & 1;
    const cpx mine = st[r], o = oth[r];
    const cpx lo = bit ? o : mine;                // amp with qubit=0
    const cpx hi = bit ? mine : o;                // amp with qubit=1
    const cpx r0 = cadd(cmul(a00, lo), cmul(a01, hi));
    const cpx r1 = cadd(cmul(a10, lo), cmul(a11, hi));
    st[r] = bit ? r1 : r0;
  }
}

// Fused XX(phix) * YY(phiy) * ZZ(phiz) on qubit-bit positions (pa, pb).
// All three couple only i <-> i^((1<<pa)|(1<<pb)); ZZ is a diagonal phase.
// With x=phix/2, y=phiy/2, z=phiz/2, e=exp(-iz):
//   equal bits   : new = cos(x-y)*e * old  +  i*sin(y-x)*e * partner
//   unequal bits : new = cos(x+y)*conj(e) * old - i*sin(x+y)*conj(e) * partner
__device__ __forceinline__ void apply_zzyyxx(cpx (&st)[8], int lane, int pa, int pb,
                                             float phiz, float phiy, float phix) {
  const float xh = 0.5f * phix, yh = 0.5f * phiy, zh = 0.5f * phiz;
  const float cz = __cosf(zh), sz = __sinf(zh);
  const cpx e  = {cz, -sz};
  const cpx ec = {cz,  sz};
  const float cdm = __cosf(xh - yh);              // eq diagonal magnitude
  const float som = __sinf(yh - xh);              // eq off-diag = i*som
  const float cdp = __cosf(xh + yh);              // neq diagonal magnitude
  const float sop = __sinf(xh + yh);              // neq off-diag = -i*sop
  const cpx dA = { cdm * e.x,   cdm * e.y };      // cos(x-y)*e
  const cpx oA = {-som * e.y,   som * e.x };      // (i*som)*e
  const cpx dB = { cdp * ec.x,  cdp * ec.y };     // cos(x+y)*ec
  const cpx oB = { sop * ec.y, -sop * ec.x };     // (-i*sop)*ec
  cpx oth[8];
  fetch_mask(st, oth, (1 << pa) | (1 << pb));
  #pragma unroll
  for (int r = 0; r < 8; ++r) {
    const int i  = (lane << 3) | r;
    const int eq = (((i >> pa) ^ (i >> pb)) & 1) == 0;
    const cpx d = eq ? dA : dB;
    const cpx o = eq ? oA : oB;
    st[r] = cadd(cmul(d, st[r]), cmul(o, oth[r]));
  }
}

__device__ __forceinline__ void apply_cu3(cpx (&st)[8], int lane, int pc, int pt,
                                          const float* __restrict__ q) {
  const float th = q[0], ph = q[1], lm = q[2];
  const float c  = __cosf(0.5f * th), sn = __sinf(0.5f * th);
  const cpx a00 = {c, 0.f};
  const cpx a01 = {-__cosf(lm) * sn, -__sinf(lm) * sn};
  const cpx a10 = { __cosf(ph) * sn,  __sinf(ph) * sn};
  const cpx a11 = { __cosf(ph + lm) * c, __sinf(ph + lm) * c};
  cpx oth[8];
  fetch_mask(st, oth, 1 << pt);                   // all lanes do the shfl
  #pragma unroll
  for (int r = 0; r < 8; ++r) {
    const int i  = (lane << 3) | r;
    const int bc = (i >> pc) & 1;
    const int bt = (i >> pt) & 1;
    const cpx mine = st[r], o = oth[r];
    const cpx lo = bt ? o : mine;
    const cpx hi = bt ? mine : o;
    const cpx r0 = cadd(cmul(a00, lo), cmul(a01, hi));
    const cpx r1 = cadd(cmul(a10, lo), cmul(a11, hi));
    const cpx nv = bt ? r1 : r0;
    st[r] = bc ? nv : mine;                       // identity when control==0
  }
}

// block: u3(a)·u3(b)·[ZZ·YY·XX fused]·u3(a)·u3(b), literal wire bit positions
#define CONV_BLOCK(A, B, WA, WB)                                           \
  do {                                                                     \
    apply_u3(st, lane, 7 - (A), (WA)[0], (WA)[1], (WA)[2]);                \
    apply_u3(st, lane, 7 - (B), (WB)[3], (WB)[4], (WB)[5]);                \
    apply_zzyyxx(st, lane, 7 - (A), 7 - (B), (WA)[6], (WA)[7], (WA)[8]);   \
    apply_u3(st, lane, 7 - (A), (WA)[9],  (WA)[10], (WA)[11]);             \
    apply_u3(st, lane, 7 - (B), (WB)[12], (WB)[13], (WB)[14]);             \
  } while (0)

__global__ void qtcn_circuit(const float* __restrict__ reduced,
                             const float* __restrict__ conv,
                             const float* __restrict__ pool,
                             float* __restrict__ expz) {
  const int lane = threadIdx.x & 31;
  // wave-uniform state index, forced into an SGPR so the angle loads below
  // become scalar (SMEM) loads
  const int s = __builtin_amdgcn_readfirstlane(
      blockIdx.x * (blockDim.x >> 5) + (threadIdx.x >> 5));
  if (s >= MROWS) return;

  // product-state init: amp(i) = prod_q (bit ? sin : cos)(theta_q / 2), real
  float cq[NQ], sq[NQ];
  #pragma unroll
  for (int q = 0; q < NQ; ++q) {
    const float h = 0.5f * reduced[s * NQ + q];   // SGPR-indexed -> s_load
    cq[q] = __cosf(h);
    sq[q] = __sinf(h);
  }
  cpx st[8];
  #pragma unroll
  for (int r = 0; r < 8; ++r) {
    const int i = (lane << 3) | r;
    float re = 1.f;
    #pragma unroll
    for (int q = 0; q < NQ; ++q)
      re *= ((i >> (7 - q)) & 1) ? sq[q] : cq[q];
    st[r] = {re, 0.f};
  }

  const float* L0 = conv;        const float* P0 = pool;
  const float* L1 = conv + 120;  const float* P1 = pool + 12;
  const float* L2 = conv + 240;  const float* P2 = pool + 24;

  // ---- layer 0: wires {0..7} ----
  CONV_BLOCK(0, 1, L0 +  0, L0 +  15);
  CONV_BLOCK(2, 3, L0 + 30, L0 +  45);
  CONV_BLOCK(4, 5, L0 + 60, L0 +  75);
  CONV_BLOCK(6, 7, L0 + 90, L0 + 105);
  CONV_BLOCK(1, 2, L0 + 15, L0 +  30);
  CONV_BLOCK(3, 4, L0 + 45, L0 +  60);
  CONV_BLOCK(5, 6, L0 + 75, L0 +  90);
  apply_cu3(st, lane, 7 - 1, 7 - 0, P0 + 0);
  apply_cu3(st, lane, 7 - 3, 7 - 2, P0 + 3);
  apply_cu3(st, lane, 7 - 5, 7 - 4, P0 + 6);
  apply_cu3(st, lane, 7 - 7, 7 - 6, P0 + 9);

  // ---- layer 1: wires {0,2,4,6} ----
  CONV_BLOCK(0, 2, L1 +  0, L1 + 15);
  CONV_BLOCK(4, 6, L1 + 30, L1 + 45);
  CONV_BLOCK(2, 4, L1 + 15, L1 + 30);
  apply_cu3(st, lane, 7 - 2, 7 - 0, P1 + 0);
  apply_cu3(st, lane, 7 - 6, 7 - 4, P1 + 3);

  // ---- layer 2: wires {0,4} ----
  CONV_BLOCK(0, 4, L2 + 0, L2 + 15);
  apply_cu3(st, lane, 7 - 4, 7 - 0, P2 + 0);

  // <Z_0>: qubit 0 is bit 7 = lane bit 4
  float ez = 0.f;
  #pragma unroll
  for (int r = 0; r < 8; ++r)
    ez += st[r].x * st[r].x + st[r].y * st[r].y;
  ez = (lane & 16) ? -ez : ez;
  #pragma unroll
  for (int off = 16; off > 0; off >>= 1)
    ez += __shfl_xor(ez, off, 32);
  if (lane == 0) expz[s] = ez;
}

// ======================= Kernel 3: per-batch mean ==========================
__global__ void qtcn_mean(const float* __restrict__ expz, float* __restrict__ out) {
  __shared__ float sm[128];
  const int b = blockIdx.x;
  float acc = 0.f;
  for (int i = threadIdx.x; i < NWIN; i += blockDim.x)
    acc += expz[b * NWIN + i];
  sm[threadIdx.x] = acc;
  __syncthreads();
  for (int sft = blockDim.x >> 1; sft > 0; sft >>= 1) {
    if ((int)threadIdx.x < sft) sm[threadIdx.x] += sm[threadIdx.x + sft];
    __syncthreads();
  }
  if (threadIdx.x == 0) out[b] = sm[0] * (1.0f / NWIN);
}

// ============================== launcher ===================================
extern "C" void kernel_launch(void* const* d_in, const int* in_sizes, int n_in,
                              void* d_out, int out_size, void* d_ws, size_t ws_size,
                              hipStream_t stream) {
  (void)in_sizes; (void)n_in; (void)out_size; (void)ws_size;
  const float* x    = (const float*)d_in[0];   // (16, 8, 384)
  const float* conv = (const float*)d_in[1];   // (3, 8, 15)
  const float* pool = (const float*)d_in[2];   // (3, 4, 3)
  const float* fcw  = (const float*)d_in[3];   // (8, 128)
  const float* fcb  = (const float*)d_in[4];   // (8,)
  float* out = (float*)d_out;                  // (16,)

  float* reduced = (float*)d_ws;               // MROWS*NQ floats  (~189 KB)
  float* expz    = reduced + MROWS * NQ;       // MROWS floats     (~24 KB)

  const dim3 blk(128);                         // 4 waves / block
  const int gemm_blocks = (MROWS / 16 + 3) / 4;      // 369 tiles -> 93 blocks
  qtcn_gemm_wmma<<<gemm_blocks, blk, 0, stream>>>(x, fcw, fcb, reduced);

  const int circ_blocks = (MROWS + 3) / 4;           // 1476 blocks, 1 wave/state
  qtcn_circuit<<<circ_blocks, blk, 0, stream>>>(reduced, conv, pool, expz);

  qtcn_mean<<<NB, blk, 0, stream>>>(expz, out);
}
